// ConvAttStage_60730837565698
// MI455X (gfx1250) — matrix operand order repaired and checked by reading, not compiled, not run on hardware
//
#include <hip/hip_runtime.h>
#include <hip/hip_bf16.h>
#include <math.h>

typedef __attribute__((ext_vector_type(16))) _Float16 v16h;
typedef __attribute__((ext_vector_type(8)))  float    v8f;

#define NBLK 2
#define BATCH 32
#define CH 384
#define NHEAD 12
#define HD 32
#define WSZ 7
#define HKW 8
#define NWIN (BATCH*HKW*HKW)   /* 2048 */
#define NTOK 50
#define HID 1536
#define QKVN (3*CH)            /* 1152 */

#define WMMA_F16(a,b,c) __builtin_amdgcn_wmma_f32_16x16x32_f16(false,(a),false,(b),(short)0,(c),false,false)

// ---- fragment loaders (CDNA5 16-bit WMMA VGPR layouts, wave32) ----
// A (16xK tile from row-major [M][ldk] f16): lane l -> row l&15, half l>>4.
// VGPR v(0..3): K = half*8 + 2v{,+1}; v(4..7): K = 16 + half*8 + ...
static __device__ __forceinline__ v16h load_a(const _Float16* p, int ld) {
  const int l = threadIdx.x & 31;
  const _Float16* q = p + (l & 15) * ld + ((l >> 4) << 3);
  v16h f;
  ((float4*)&f)[0] = ((const float4*)q)[0];
  ((float4*)&f)[1] = ((const float4*)(q + 16))[0];
  return f;
}
// B (KxN tile, sourced from row-major [N][ldk] f16 = W rows for x@W^T):
// lane l -> col l&15, half l>>4: K = half*8 + 0..7, then 16 + half*8 + 0..7.
static __device__ __forceinline__ v16h load_b(const _Float16* p, int ld) {
  const int l = threadIdx.x & 31;
  const _Float16* q = p + (l & 15) * ld + ((l >> 4) << 3);
  v16h f;
  ((float4*)&f)[0] = ((const float4*)q)[0];
  ((float4*)&f)[1] = ((const float4*)(q + 16))[0];
  return f;
}
// D tile (f32 acc) -> f16 row-major store: lane l holds N=l&15, M = v + 8*(l>>4).
static __device__ __forceinline__ void store_d_f16(_Float16* p, int ld, v8f d) {
  const int l = threadIdx.x & 31;
  const int n = l & 15, mh = (l >> 4) << 3;
#pragma unroll
  for (int v = 0; v < 8; ++v) p[(mh + v) * ld + n] = (_Float16)d[v];
}

// ------------------- layout pack / unpack -------------------
__global__ void pack_kernel(const float* __restrict__ cls, const float* __restrict__ patch,
                            float* __restrict__ X) {
  size_t idx = (size_t)blockIdx.x * blockDim.x + threadIdx.x;
  const size_t total = (size_t)NWIN * NTOK * CH;
  if (idx >= total) return;
  int c = (int)(idx % CH);
  int pos = (int)((idx / CH) % NTOK);
  int win = (int)(idx / ((size_t)CH * NTOK));
  int b = win >> 6, wloc = win & 63, hy = wloc >> 3, hx = wloc & 7;
  float v;
  if (pos == 0) {
    v = cls[(((size_t)b * CH + c) * HKW + hy) * HKW + hx];
  } else {
    int pp = pos - 1, wy = pp / WSZ, wx = pp % WSZ;
    int y = hy * WSZ + wy, x = hx * WSZ + wx;
    v = patch[(((size_t)b * CH + c) * 56 + y) * 56 + x];
  }
  X[idx] = v;
}

__global__ void unpack_kernel(const float* __restrict__ X, float* __restrict__ out) {
  size_t idx = (size_t)blockIdx.x * blockDim.x + threadIdx.x;
  const size_t total = (size_t)NWIN * NTOK * CH;
  if (idx >= total) return;
  int c = (int)(idx % CH);
  int pos = (int)((idx / CH) % NTOK);
  int win = (int)(idx / ((size_t)CH * NTOK));
  int b = win >> 6, wloc = win & 63, hy = wloc >> 3, hx = wloc & 7;
  float v = X[idx];
  if (pos == 0) {
    out[(((size_t)b * CH + c) * HKW + hy) * HKW + hx] = v;
  } else {
    int pp = pos - 1, wy = pp / WSZ, wx = pp % WSZ;
    int y = hy * WSZ + wy, x = hx * WSZ + wx;
    out[(size_t)BATCH * CH * HKW * HKW + (((size_t)b * CH + c) * 56 + y) * 56 + x] = v;
  }
}

__global__ void f32_to_f16_kernel(const float* __restrict__ s, _Float16* __restrict__ d, int n) {
  int i = blockIdx.x * blockDim.x + threadIdx.x;
  if (i < n) d[i] = (_Float16)s[i];
}

// padded + masked relative-position bias table: [NHEAD][64][64]
__global__ void rpb_kernel(const float* __restrict__ rel_pos, float* __restrict__ rpb, int blk) {
  int idx = blockIdx.x * blockDim.x + threadIdx.x;
  int n = idx & 63, r = (idx >> 6) & 63, h = idx >> 12;
  float v;
  if (n >= NTOK) v = -1e30f;                       // mask padding cols
  else if (r == 0 || n == 0 || r >= NTOK) v = 0.f; // zero bias on cls row/col
  else {
    int qi = r - 1, ki = n - 1;
    int qy = qi / WSZ, qx = qi % WSZ, ky = ki / WSZ, kx = ki % WSZ;
    const int rd = 2 * WSZ - 1;
    int dy = ((ky - qy + WSZ / 2) % rd + rd) % rd;
    int dx = ((kx - qx + WSZ / 2) % rd + rd) % rd;
    v = rel_pos[(size_t)(blk * NHEAD + h) * (rd * rd) + dy * rd + dx];
  }
  rpb[idx] = v;
}

// ------------------- fused attention (cls or window) -------------------
__global__ __launch_bounds__(256) void attn_kernel(
    float* __restrict__ X,
    const _Float16* __restrict__ wqkv,   // [1152][384] f16
    const _Float16* __restrict__ wproj,  // [384][384] f16
    const float* __restrict__ bqkv, const float* __restrict__ bproj,
    const float* __restrict__ lnw, const float* __restrict__ lnb,
    const float* __restrict__ rpb,       // [12][64][64] or nullptr
    int mode)                            // 0 = cls attn, 1 = window attn
{
  __shared__ _Float16 xln[64 * CH];      // 48 KB : LN'd inputs (f16)
  __shared__ _Float16 attnout[64 * CH];  // 48 KB : attention output (f16)
  __shared__ _Float16 qh[64 * HD];       // per-head Q
  __shared__ _Float16 kh[64 * HD];       // per-head K
  __shared__ _Float16 vT[HD * 64];       // per-head V, transposed
  __shared__ _Float16 pbuf[64 * 64];     // per-head softmax P

  int nvalid, rstride; size_t base;
  if (mode == 0) { nvalid = 64;   rstride = NTOK * CH; base = (size_t)blockIdx.x * 64 * NTOK * CH; }
  else           { nvalid = NTOK; rstride = CH;        base = (size_t)blockIdx.x * NTOK * CH; }

  // ---- LayerNorm -> xln (f16); pad rows zeroed ----
  {
    int t = threadIdx.x;
    int r = t >> 2, qd = t & 3;
    const float* xr = X + base + (size_t)r * rstride;
    float s = 0.f, ss = 0.f;
    if (r < nvalid) {
      for (int c = qd * 96; c < qd * 96 + 96; ++c) { float v = xr[c]; s += v; ss += v * v; }
    }
    s  += __shfl_xor(s, 1, 32);  s  += __shfl_xor(s, 2, 32);
    ss += __shfl_xor(ss, 1, 32); ss += __shfl_xor(ss, 2, 32);
    float mean = s * (1.f / CH);
    float inv = rsqrtf(ss * (1.f / CH) - mean * mean + 1e-6f);
    for (int c = qd * 96; c < qd * 96 + 96; ++c) {
      float o = 0.f;
      if (r < nvalid) o = (xr[c] - mean) * inv * lnw[c] + lnb[c];
      xln[r * CH + c] = (_Float16)o;
    }
  }
  __syncthreads();

  const int wv = threadIdx.x >> 5;
  const int l = threadIdx.x & 31;
  const int n16 = l & 15, mh = (l >> 4) << 3;
  const float scale = 0.17677669529663687f; // 1/sqrt(32)

  for (int h = 0; h < NHEAD; ++h) {
    // ---- per-head QKV GEMM: 24 output tiles across 8 waves ----
#pragma unroll
    for (int j = 0; j < 3; ++j) {
      int tid = wv * 3 + j;
      int mt = tid & 3, p = tid >> 2;    // p: 0..5 -> {q0,q1,k0,k1,v0,v1}
      int seg = p >> 1, hp = p & 1;
      int colg = seg * CH + h * HD + hp * 16;
      v8f acc = {};
      for (int kk = 0; kk < CH / 32; ++kk) {
        v16h a = load_a(xln + mt * 16 * CH + kk * 32, CH);
        v16h b = load_b(wqkv + (size_t)colg * CH + kk * 32, CH);
        acc = WMMA_F16(a, b, acc);
      }
      float bias = bqkv[colg + n16];
      if (seg < 2) {
        _Float16* dst = (seg == 0) ? qh : kh;
#pragma unroll
        for (int v = 0; v < 8; ++v)
          dst[(mt * 16 + mh + v) * HD + hp * 16 + n16] = (_Float16)(acc[v] + bias);
      } else {
#pragma unroll
        for (int v = 0; v < 8; ++v)
          vT[(hp * 16 + n16) * 64 + (mt * 16 + mh + v)] = (_Float16)(acc[v] + bias);
      }
    }
    __syncthreads();

    // ---- scores + softmax: waves 0..3, one 16-row block each ----
    if (wv < 4) {
      int mt = wv;
      v8f S[4];
      v16h a = load_a(qh + mt * 16 * HD, HD);
#pragma unroll
      for (int nt = 0; nt < 4; ++nt) {
        v16h b = load_b(kh + nt * 16 * HD, HD);
        v8f z = {};
        S[nt] = WMMA_F16(a, b, z);
      }
#pragma unroll
      for (int nt = 0; nt < 4; ++nt)
#pragma unroll
        for (int v = 0; v < 8; ++v) {
          float x = S[nt][v] * scale;
          if (rpb) x += rpb[(h * 64 + (mt * 16 + mh + v)) * 64 + nt * 16 + n16];
          S[nt][v] = x;
        }
#pragma unroll
      for (int v = 0; v < 8; ++v) {
        float mx = fmaxf(fmaxf(S[0][v], S[1][v]), fmaxf(S[2][v], S[3][v]));
        for (int o = 8; o >= 1; o >>= 1) mx = fmaxf(mx, __shfl_xor(mx, o, 32));
        float sum = 0.f;
#pragma unroll
        for (int nt = 0; nt < 4; ++nt) { float e = __expf(S[nt][v] - mx); S[nt][v] = e; sum += e; }
        for (int o = 8; o >= 1; o >>= 1) sum += __shfl_xor(sum, o, 32);
        float rs = 1.f / sum;
#pragma unroll
        for (int nt = 0; nt < 4; ++nt) S[nt][v] *= rs;
      }
#pragma unroll
      for (int nt = 0; nt < 4; ++nt)
        store_d_f16(pbuf + mt * 16 * 64 + nt * 16, 64, S[nt]);
    }
    __syncthreads();

    // ---- O = P @ V : 8 tiles, one per wave ----
    {
      int mt = wv & 3, ot = wv >> 2;
      v8f acc = {};
#pragma unroll
      for (int kt = 0; kt < 2; ++kt) {
        v16h a = load_a(pbuf + mt * 16 * 64 + kt * 32, 64);
        v16h b = load_b(vT + ot * 16 * 64 + kt * 32, 64);
        acc = WMMA_F16(a, b, acc);
      }
      store_d_f16(attnout + mt * 16 * CH + h * HD + ot * 16, CH, acc);
    }
    __syncthreads();
  }

  // ---- projection + residual add into X ----
  for (int id = wv; id < 96; id += 8) {
    int mt = id & 3, nt = id >> 2;
    v8f acc = {};
    for (int kk = 0; kk < CH / 32; ++kk) {
      v16h a = load_a(attnout + mt * 16 * CH + kk * 32, CH);
      v16h b = load_b(wproj + (size_t)(nt * 16) * CH + kk * 32, CH);
      acc = WMMA_F16(a, b, acc);
    }
    int coln = nt * 16 + n16;
    float pb = bproj[coln];
#pragma unroll
    for (int v = 0; v < 8; ++v) {
      int m = mt * 16 + mh + v;
      if (m < nvalid) X[base + (size_t)m * rstride + coln] += acc[v] + pb;
    }
  }
}

// ------------------- fused LN2 + MLP (64 tokens / workgroup) -------------------
__global__ __launch_bounds__(256) void mlp_kernel(
    float* __restrict__ X,
    const _Float16* __restrict__ w1,  // [1536][384]
    const _Float16* __restrict__ w2,  // [384][1536]
    const float* __restrict__ b1, const float* __restrict__ b2,
    const float* __restrict__ lnw, const float* __restrict__ lnb)
{
  __shared__ _Float16 xln[64 * CH];    // 48 KB
  __shared__ _Float16 hbuf[64 * 768];  // 96 KB (half of hidden at a time)

  size_t base = (size_t)blockIdx.x * 64 * CH;

  // LN2 -> xln
  {
    int t = threadIdx.x;
    int r = t >> 2, qd = t & 3;
    const float* xr = X + base + (size_t)r * CH;
    float s = 0.f, ss = 0.f;
    for (int c = qd * 96; c < qd * 96 + 96; ++c) { float v = xr[c]; s += v; ss += v * v; }
    s  += __shfl_xor(s, 1, 32);  s  += __shfl_xor(s, 2, 32);
    ss += __shfl_xor(ss, 1, 32); ss += __shfl_xor(ss, 2, 32);
    float mean = s * (1.f / CH);
    float inv = rsqrtf(ss * (1.f / CH) - mean * mean + 1e-6f);
    for (int c = qd * 96; c < qd * 96 + 96; ++c)
      xln[r * CH + c] = (_Float16)((xr[c] - mean) * inv * lnw[c] + lnb[c]);
  }
  __syncthreads();

  const int wv = threadIdx.x >> 5;
  const int l = threadIdx.x & 31;
  const int n16 = l & 15, mh = (l >> 4) << 3;

  v8f out[12] = {};

  for (int p = 0; p < 2; ++p) {
    // phase 1: h = GELU(xln @ W1^T + b1) for this hidden half
    for (int id = wv; id < 192; id += 8) {
      int mt = id & 3, nt = id >> 2;
      int col = p * 768 + nt * 16;
      v8f acc = {};
      for (int kk = 0; kk < CH / 32; ++kk) {
        v16h a = load_a(xln + mt * 16 * CH + kk * 32, CH);
        v16h b = load_b(w1 + (size_t)col * CH + kk * 32, CH);
        acc = WMMA_F16(a, b, acc);
      }
      float bb = b1[col + n16];
#pragma unroll
      for (int v = 0; v < 8; ++v) {
        float x = acc[v] + bb;
        float g = 0.5f * x * (1.0f + erff(x * 0.70710678118654752f)); // exact gelu
        hbuf[(mt * 16 + mh + v) * 768 + nt * 16 + n16] = (_Float16)g;
      }
    }
    __syncthreads();
    // phase 2: out += h @ W2^T (this hidden half), accumulators in registers
#pragma unroll
    for (int j = 0; j < 12; ++j) {
      int mt = j & 3, nt = wv * 3 + (j >> 2);
      v8f acc = out[j];
      for (int kk = 0; kk < 24; ++kk) {
        v16h a = load_a(hbuf + mt * 16 * 768 + kk * 32, 768);
        v16h b = load_b(w2 + (size_t)(nt * 16) * HID + p * 768 + kk * 32, HID);
        acc = WMMA_F16(a, b, acc);
      }
      out[j] = acc;
    }
    __syncthreads();
  }

  // residual add
#pragma unroll
  for (int j = 0; j < 12; ++j) {
    int mt = j & 3, nt = wv * 3 + (j >> 2);
    int col = nt * 16 + n16;
    float bb = b2[col];
#pragma unroll
    for (int v = 0; v < 8; ++v)
      X[base + (size_t)(mt * 16 + mh + v) * CH + col] += out[j][v] + bb;
  }
}

extern "C" void kernel_launch(void* const* d_in, const int* in_sizes, int n_in,
                              void* d_out, int out_size, void* d_ws, size_t ws_size,
                              hipStream_t stream) {
  const float* cls     = (const float*)d_in[0];
  const float* patch   = (const float*)d_in[1];
  const float* qkv_w   = (const float*)d_in[2];
  const float* qkv_b   = (const float*)d_in[3];
  const float* proj_w  = (const float*)d_in[4];
  const float* proj_b  = (const float*)d_in[5];
  const float* rel_pos = (const float*)d_in[6];
  const float* ln0_w = (const float*)d_in[7];  const float* ln0_b = (const float*)d_in[8];
  const float* ln1_w = (const float*)d_in[9];  const float* ln1_b = (const float*)d_in[10];
  const float* ln2_w = (const float*)d_in[11]; const float* ln2_b = (const float*)d_in[12];
  const float* mlp_w1 = (const float*)d_in[13]; const float* mlp_b1 = (const float*)d_in[14];
  const float* mlp_w2 = (const float*)d_in[15]; const float* mlp_b2 = (const float*)d_in[16];

  char* ws = (char*)d_ws;
  size_t off = 0;
  float*    X      = (float*)(ws + off);    off += (size_t)NWIN * NTOK * CH * 4;   // 157 MB residual stream
  _Float16* qkv16  = (_Float16*)(ws + off); off += (size_t)NBLK * QKVN * CH * 2;
  _Float16* proj16 = (_Float16*)(ws + off); off += (size_t)NBLK * CH * CH * 2;
  _Float16* w116   = (_Float16*)(ws + off); off += (size_t)NBLK * HID * CH * 2;
  _Float16* w216   = (_Float16*)(ws + off); off += (size_t)NBLK * CH * HID * 2;
  float*    rpb    = (float*)(ws + off);    off += (size_t)NHEAD * 64 * 64 * 4;

  const size_t tot = (size_t)NWIN * NTOK * CH;
  pack_kernel<<<(unsigned)((tot + 255) / 256), 256, 0, stream>>>(cls, patch, X);
  f32_to_f16_kernel<<<(NBLK * QKVN * CH + 255) / 256, 256, 0, stream>>>(qkv_w, qkv16, NBLK * QKVN * CH);
  f32_to_f16_kernel<<<(NBLK * CH * CH + 255) / 256, 256, 0, stream>>>(proj_w, proj16, NBLK * CH * CH);
  f32_to_f16_kernel<<<(NBLK * HID * CH + 255) / 256, 256, 0, stream>>>(mlp_w1, w116, NBLK * HID * CH);
  f32_to_f16_kernel<<<(NBLK * CH * HID + 255) / 256, 256, 0, stream>>>(mlp_w2, w216, NBLK * CH * HID);

  for (int i = 0; i < NBLK; ++i) {
    // cls (region) attention: 32 workgroups, 64 tokens each
    attn_kernel<<<BATCH, 256, 0, stream>>>(
        X, qkv16 + (size_t)i * QKVN * CH, proj16 + (size_t)i * CH * CH,
        qkv_b + i * QKVN, proj_b + i * CH, ln0_w + i * CH, ln0_b + i * CH,
        nullptr, 0);
    // rel-pos bias table for this block
    rpb_kernel<<<(NHEAD * 64 * 64) / 256, 256, 0, stream>>>(rel_pos, rpb, i);
    // window attention: 2048 workgroups
    attn_kernel<<<NWIN, 256, 0, stream>>>(
        X, qkv16 + (size_t)i * QKVN * CH, proj16 + (size_t)i * CH * CH,
        qkv_b + i * QKVN, proj_b + i * CH, ln1_w + i * CH, ln1_b + i * CH,
        rpb, 1);
    // MLP: 1600 workgroups of 64 tokens
    mlp_kernel<<<(NWIN * NTOK) / 64, 256, 0, stream>>>(
        X, w116 + (size_t)i * HID * CH, w216 + (size_t)i * CH * HID,
        mlp_b1 + i * HID, mlp_b2 + i * CH, ln2_w + i * CH, ln2_b + i * CH);
  }

  unpack_kernel<<<(unsigned)((tot + 255) / 256), 256, 0, stream>>>(X, (float*)d_out);
}
